// XConv_7765300871438
// MI455X (gfx1250) — matrix operand order repaired and checked by reading, not compiled, not run on hardware
//
#include <hip/hip_runtime.h>
#include <hip/hip_bf16.h>
#include <cstring>

#define B_    8
#define N_    8192
#define M_    2048
#define K_    16
#define CIN   64
#define COUT  128
#define MID   32
#define K2    256
#define CF    96            // MID + CIN
#define RED   1536          // CF * K_
#define NPTS  (B_ * M_)     // 16384
#define EPSN  1e-5f
#define QSZ   (B_ * M_ * 3) // 49152

typedef __attribute__((ext_vector_type(16))) __bf16 v16bf;
typedef __attribute__((ext_vector_type(8)))  __bf16 v8bf;
typedef __attribute__((ext_vector_type(8)))  float  v8f;
typedef unsigned int u32x4 __attribute__((ext_vector_type(4)));
typedef int          i32x4 __attribute__((ext_vector_type(4)));
typedef int          i32x8 __attribute__((ext_vector_type(8)));

__device__ __forceinline__ v8f wmma_bf16(v16bf a, v16bf b, v8f c) {
  // v_wmma_f32_16x16x32_bf16: D = A(16x32) * B(32x16) + C
  return __builtin_amdgcn_wmma_f32_16x16x32_bf16(false, a, false, b, (short)0, c,
                                                 false, false);
}

// ---- Tensor Data Mover: DMA a dense 2D tile [rows x width] into LDS --------
// D# bitfields per CDNA5 ISA §8 (async_tensor). elem_log2: 0=1B,1=2B,2=4B.
__device__ __forceinline__ void tdm_load_2d(unsigned lds_off, const void* gptr,
                                            unsigned elem_log2, unsigned width,
                                            unsigned rows, unsigned stride) {
  const unsigned long long ga = (unsigned long long)gptr;
  u32x4 g0;
  g0[0] = 1u;                                            // count=1, user D#
  g0[1] = lds_off;                                       // lds_addr (bytes)
  g0[2] = (unsigned)(ga & 0xFFFFFFFFu);                  // global_addr lo
  g0[3] = (unsigned)((ga >> 32) & 0x1FFFFFFu) | (2u << 30); // addr hi | type=2
  i32x8 g1;
  g1[0] = (int)(elem_log2 << 16);                        // data_size
  g1[1] = (int)((width & 0xFFFFu) << 16);                // tensor_dim0[15:0]
  g1[2] = (int)((width >> 16) & 0xFFFFu) | (int)((rows & 0xFFFFu) << 16); // dim0 hi | tensor_dim1 lo
  g1[3] = (int)((rows >> 16) & 0xFFFFu) | (int)((width & 0xFFFFu) << 16); // dim1 hi | tile_dim0
  g1[4] = (int)(rows & 0xFFFFu);                         // tile_dim1 (tile_dim2=0)
  g1[5] = (int)stride;                                   // tensor_dim0_stride lo32
  g1[6] = 0;                                             // stride hi | dim1_stride lo
  g1[7] = 0;
  i32x4 z4 = {0, 0, 0, 0};
#if defined(__clang_major__) && __clang_major__ >= 23
  i32x8 z8 = {0, 0, 0, 0, 0, 0, 0, 0};
  __builtin_amdgcn_tensor_load_to_lds(g0, g1, z4, z4, z8, 0);
#else
  __builtin_amdgcn_tensor_load_to_lds(g0, g1, z4, z4, 0);
#endif
}

struct InPack {
  const float *p, *q, *x;
  const float *mw1, *mb1, *m1g, *m1b, *m1m, *m1v;
  const float *mw2, *mb2, *m2g, *m2b, *m2m, *m2v;
  const float *sw1, *sb1, *s1g, *s1b, *s1m, *s1v;
  const float *sw2, *sb2, *s2g, *s2b, *s2m, *s2v;
  const float *sw3, *sb3;
  const float *cw, *cbias, *cg, *cb, *cm, *cv;
};

// ---------------------------------------------------------------- utilities
__global__ void cvt_bf16_kernel(const float* __restrict__ in,
                                __bf16* __restrict__ out, int n) {
  int i = blockIdx.x * blockDim.x + threadIdx.x;
  if (i < n) out[i] = (__bf16)in[i];
}

__global__ void copyq_kernel(const float* __restrict__ q, float* __restrict__ out) {
  int i = blockIdx.x * blockDim.x + threadIdx.x;
  if (i < QSZ) out[i] = q[i];
}

// ---------------------------------------------------------------- 1) KNN
#define TP 512
__global__ __launch_bounds__(256) void knn_kernel(const float* __restrict__ p,
                                                  const float* __restrict__ q,
                                                  int* __restrict__ idx_out) {
  const int b = blockIdx.y;
  const int m = blockIdx.x * blockDim.x + threadIdx.x;
  const float* pb = p + (long)b * N_ * 3;

  const float qx = q[((long)b * M_ + m) * 3 + 0];
  const float qy = q[((long)b * M_ + m) * 3 + 1];
  const float qz = q[((long)b * M_ + m) * 3 + 2];

  float dk[K_];
  int   ik[K_];
#pragma unroll
  for (int t = 0; t < K_; ++t) { dk[t] = 3.4e38f; ik[t] = 0; }

  __shared__ float sp[TP * 3];

  for (int t0 = 0; t0 < N_; t0 += TP) {
    __syncthreads();
    for (int i = threadIdx.x; i < TP * 3; i += blockDim.x) sp[i] = pb[t0 * 3 + i];
    if (t0 + TP < N_)  // hint next tile into L2 (global_prefetch_b8)
      __builtin_prefetch(&pb[(t0 + TP) * 3 + threadIdx.x * 6], 0, 1);
    __syncthreads();

    for (int j = 0; j < TP; ++j) {
      const float dx = sp[j * 3 + 0] - qx;
      const float dy = sp[j * 3 + 1] - qy;
      const float dz = sp[j * 3 + 2] - qz;
      const float d2 = dx * dx + dy * dy + dz * dz;
      if (d2 < dk[K_ - 1]) {
        dk[K_ - 1] = d2; ik[K_ - 1] = t0 + j;
#pragma unroll
        for (int t = K_ - 1; t > 0; --t) {
          if (dk[t] < dk[t - 1]) {
            float td = dk[t]; dk[t] = dk[t - 1]; dk[t - 1] = td;
            int   ti = ik[t]; ik[t] = ik[t - 1]; ik[t - 1] = ti;
          }
        }
      }
    }
  }
#pragma unroll
  for (int t = 0; t < K_; ++t)
    idx_out[((long)b * M_ + m) * K_ + t] = ik[t];
}

// -------------------------------------- 2) gather + p_hat + MLP + STN layer1
__global__ __launch_bounds__(256) void feat_kernel(InPack P,
                                                   const int* __restrict__ idx,
                                                   __bf16* __restrict__ F,
                                                   __bf16* __restrict__ t1) {
  const int lane = threadIdx.x & 31;
  const int w    = threadIdx.x >> 5;          // wave in block = point slot
  const int pt   = blockIdx.x * 8 + w;
  const int b    = pt / M_;
  const int m    = pt - b * M_;
  const int k    = lane & 15;
  const int half = lane >> 4;

  __shared__ float ph[8][K_][3];

  const int nb = idx[(long)pt * K_ + k];
  const float qx = P.q[((long)b * M_ + m) * 3 + 0];
  const float qy = P.q[((long)b * M_ + m) * 3 + 1];
  const float qz = P.q[((long)b * M_ + m) * 3 + 2];
  const float px = P.p[((long)b * N_ + nb) * 3 + 0] - qx;
  const float py = P.p[((long)b * N_ + nb) * 3 + 1] - qy;
  const float pz = P.p[((long)b * N_ + nb) * 3 + 2] - qz;

  if (half == 0) {
    ph[w][k][0] = px; ph[w][k][1] = py; ph[w][k][2] = pz;
    // coords MLP: 3 -> 32 -> 32, bn+relu, write into F[:, k, 0:32]
    float h1[MID];
#pragma unroll
    for (int c = 0; c < MID; ++c) {
      float v = P.mb1[c] + P.mw1[c * 3 + 0] * px + P.mw1[c * 3 + 1] * py +
                P.mw1[c * 3 + 2] * pz;
      const float sc = P.m1g[c] * rsqrtf(P.m1v[c] + EPSN);
      v = (v - P.m1m[c]) * sc + P.m1b[c];
      h1[c] = fmaxf(v, 0.f);
    }
#pragma unroll 4
    for (int o = 0; o < MID; ++o) {
      float v = P.mb2[o];
#pragma unroll
      for (int c = 0; c < MID; ++c) v += P.mw2[o * MID + c] * h1[c];
      const float sc = P.m2g[o] * rsqrtf(P.m2v[o] + EPSN);
      v = (v - P.m2m[o]) * sc + P.m2b[o];
      F[(long)pt * RED + k * CF + o] = (__bf16)fmaxf(v, 0.f);
    }
  } else {
    // gather input features x[b, :, nb] into F[:, k, 32:96]
#pragma unroll 8
    for (int c = 0; c < CIN; ++c)
      F[(long)pt * RED + k * CF + MID + c] =
          (__bf16)P.x[((long)b * CIN + c) * N_ + nb];
  }
  __syncthreads();

  // STN layer 1: t1[o] = relu(bn(sum_{c,k} sw1[o,c,k] * ph[k][c] + sb1[o]))
#pragma unroll
  for (int oo = 0; oo < 8; ++oo) {
    const int o = lane * 8 + oo;
    float acc = P.sb1[o];
#pragma unroll
    for (int c = 0; c < 3; ++c)
#pragma unroll
      for (int kk = 0; kk < K_; ++kk)
        acc += P.sw1[o * 48 + c * K_ + kk] * ph[w][kk][c];
    const float sc = P.s1g[o] * rsqrtf(P.s1v[o] + EPSN);
    acc = (acc - P.s1m[o]) * sc + P.s1b[o];
    t1[(long)pt * K2 + o] = (__bf16)fmaxf(acc, 0.f);
  }
}

// -------------------------------------- 3) STN layers 2/3: [NPTS,256]x[256,256]
// All-bf16: A tile (16x256 bf16 = 8KB) DMA'd into LDS via TDM; B pre-converted.
__global__ __launch_bounds__(128) void gemm256_kernel(
    const __bf16* __restrict__ A, const __bf16* __restrict__ Wt,
    const float* __restrict__ bias, const float* __restrict__ g,
    const float* __restrict__ bb, const float* __restrict__ mm,
    const float* __restrict__ vv, int do_bnrelu, __bf16* __restrict__ out) {
  const int lane = threadIdx.x & 31;
  const int wv   = threadIdx.x >> 5;         // 0..3: 64 output cols each
  const int row0 = blockIdx.x * 16;
  const int col0 = wv * 64;
  const int r    = lane & 15;
  const int hh   = lane >> 4;

  __shared__ __align__(16) __bf16 sA[16 * K2];   // 8KB A tile
  if (wv == 0) {
    tdm_load_2d((unsigned)(unsigned long long)(void*)sA,
                A + (long)row0 * K2, /*elem_log2=*/1, /*width=*/K2,
                /*rows=*/16, /*stride=*/K2);
    __builtin_amdgcn_s_wait_tensorcnt(0);
  }
  __syncthreads();

  v8f acc[4] = {};
#pragma unroll 2
  for (int s = 0; s < 8; ++s) {
    const int k0 = s * 32;
    const v8bf alo = *(const v8bf*)(sA + r * K2 + k0 + hh * 8);      // ds_load_b128
    const v8bf ahi = *(const v8bf*)(sA + r * K2 + k0 + hh * 8 + 16);
    v16bf a;
#pragma unroll
    for (int j = 0; j < 8; ++j) { a[j] = alo[j]; a[8 + j] = ahi[j]; }
#pragma unroll
    for (int t = 0; t < 4; ++t) {
      const int col = col0 + t * 16 + r;
      const __bf16* wrow = Wt + (long)col * K2 + k0 + hh * 8;
      const v8bf blo = *(const v8bf*)(wrow);                         // global_load_b128
      const v8bf bhi = *(const v8bf*)(wrow + 16);
      v16bf bm;
#pragma unroll
      for (int j = 0; j < 8; ++j) { bm[j] = blo[j]; bm[8 + j] = bhi[j]; }
      acc[t] = wmma_bf16(a, bm, acc[t]);
    }
  }
#pragma unroll
  for (int t = 0; t < 4; ++t) {
    const int col = col0 + t * 16 + r;
    const float bv = bias[col];
    float sc = 1.f, sh = 0.f;
    if (do_bnrelu) {
      sc = g[col] * rsqrtf(vv[col] + EPSN);
      sh = bb[col] - mm[col] * sc;
    }
#pragma unroll
    for (int rr = 0; rr < 8; ++rr) {
      const int rowi = row0 + rr + hh * 8;
      float v = acc[t][rr] + bv;
      if (do_bnrelu) v = fmaxf(v * sc + sh, 0.f);
      out[(long)rowi * K2 + col] = (__bf16)v;
    }
  }
}

// -------------------------------------- 4) X-transform: per point T(16x16) x F(16x96)
__global__ __launch_bounds__(256) void xform_kernel(const __bf16* __restrict__ t3,
                                                    const __bf16* __restrict__ F,
                                                    __bf16* __restrict__ Y) {
  const int lane = threadIdx.x & 31;
  const int w    = threadIdx.x >> 5;
  const int pt   = blockIdx.x * 8 + w;
  const int r    = lane & 15;
  const int hh   = lane >> 4;

  // A = T zero-padded 16x32 (K 16..31 = 0). T[i][j] = t3[pt*256 + i*16 + j]
  const v8bf tr = *(const v8bf*)(t3 + (long)pt * K2 + r * 16 + hh * 8);
  v16bf a;
#pragma unroll
  for (int j = 0; j < 8; ++j) { a[j] = tr[j]; a[8 + j] = (__bf16)0.0f; }

#pragma unroll
  for (int t = 0; t < 6; ++t) {          // 6 column tiles of 16 over 96 channels
    v16bf bm;
    {                                     // K rows 0..15 carry F; 16..31 padded 0
      const __bf16* fc = F + (long)pt * RED + (t * 16 + r);
#pragma unroll
      for (int j = 0; j < 8; ++j) { bm[j] = fc[(long)(hh * 8 + j) * CF]; bm[8 + j] = (__bf16)0.0f; }
    }
    v8f acc = {};
    acc = wmma_bf16(a, bm, acc);
    // store y[pt][c][i] (c-major so the final GEMM reduction dim is contiguous)
#pragma unroll
    for (int rr = 0; rr < 8; ++rr) {
      const int i = rr + hh * 8;
      const int c = t * 16 + r;
      Y[(long)pt * RED + c * K_ + i] = (__bf16)acc[rr];
    }
  }
}

// -------------------------------------- 5) final conv: [NPTS,1536]x[1536,128]
// A tile (16x1536 bf16 = 48KB) DMA'd into LDS once per block via TDM.
__global__ __launch_bounds__(128) void final_kernel(const __bf16* __restrict__ Y,
                                                    const __bf16* __restrict__ Wb,
                                                    InPack P,
                                                    float* __restrict__ out) {
  const int lane = threadIdx.x & 31;
  const int wv   = threadIdx.x >> 5;     // 0..3: 32 cols each
  const int row0 = blockIdx.x * 16;
  const int col0 = wv * 32;
  const int r    = lane & 15;
  const int hh   = lane >> 4;

  __shared__ __align__(16) __bf16 sA[16 * RED];   // 48KB A tile
  if (wv == 0) {
    tdm_load_2d((unsigned)(unsigned long long)(void*)sA,
                Y + (long)row0 * RED, /*elem_log2=*/1, /*width=*/RED,
                /*rows=*/16, /*stride=*/RED);
    __builtin_amdgcn_s_wait_tensorcnt(0);
  }
  __syncthreads();

  v8f acc[2] = {};
  for (int s = 0; s < 48; ++s) {
    const int k0 = s * 32;
    const v8bf alo = *(const v8bf*)(sA + r * RED + k0 + hh * 8);
    const v8bf ahi = *(const v8bf*)(sA + r * RED + k0 + hh * 8 + 16);
    v16bf a;
#pragma unroll
    for (int j = 0; j < 8; ++j) { a[j] = alo[j]; a[8 + j] = ahi[j]; }
#pragma unroll
    for (int t = 0; t < 2; ++t) {
      const int col = col0 + t * 16 + r;
      const __bf16* wrow = Wb + (long)col * RED + k0 + hh * 8;
      const v8bf blo = *(const v8bf*)(wrow);      // global_load_b128
      const v8bf bhi = *(const v8bf*)(wrow + 16);
      v16bf bm;
#pragma unroll
      for (int j = 0; j < 8; ++j) { bm[j] = blo[j]; bm[8 + j] = bhi[j]; }
      acc[t] = wmma_bf16(a, bm, acc[t]);
    }
  }
#pragma unroll
  for (int t = 0; t < 2; ++t) {
    const int col = col0 + t * 16 + r;
    const float bv = P.cbias[col];
    const float sc = P.cg[col] * rsqrtf(P.cv[col] + EPSN);
    const float sh = P.cb[col] - P.cm[col] * sc;
#pragma unroll
    for (int rr = 0; rr < 8; ++rr) {
      const int pt = row0 + rr + hh * 8;
      const int b  = pt / M_;
      const int m  = pt - b * M_;
      const float v = fmaxf((acc[t][rr] + bv) * sc + sh, 0.f);
      out[QSZ + ((long)b * COUT + col) * M_ + m] = v;
    }
  }
}

// ---------------------------------------------------------------- launcher
extern "C" void kernel_launch(void* const* d_in, const int* in_sizes, int n_in,
                              void* d_out, int out_size, void* d_ws, size_t ws_size,
                              hipStream_t stream) {
  (void)in_sizes; (void)out_size; (void)ws_size;
  const float* fp[35];
  for (int i = 0; i < 35 && i < n_in; ++i) fp[i] = (const float*)d_in[i];
  InPack P;
  memcpy(&P, fp, sizeof(P));   // struct fields match setup_inputs() flat order

  char* ws = (char*)d_ws;
  int*    idx = (int*)(ws);                        //  1 MB
  __bf16* t1  = (__bf16*)(ws + (1ull   << 20));    //  8 MB (bf16)
  __bf16* t2  = (__bf16*)(ws + (9ull   << 20));    //  8 MB (bf16)
  __bf16* t3  = t1;                                // reuse t1 region
  __bf16* F   = (__bf16*)(ws + (17ull  << 20));    // 48 MB
  __bf16* Y   = (__bf16*)(ws + (65ull  << 20));    // 48 MB
  __bf16* Wbc = (__bf16*)(ws + (113ull << 20));    // conv_w bf16, 384 KB
  __bf16* Wb2 = (__bf16*)(ws + (114ull << 20));    // stn_w2 bf16, 128 KB
  __bf16* Wb3 = (__bf16*)(ws + (115ull << 20));    // stn_w3 bf16, 128 KB

  cvt_bf16_kernel<<<(COUT * RED + 255) / 256, 256, 0, stream>>>(P.cw, Wbc, COUT * RED);
  cvt_bf16_kernel<<<(K2 * K2 + 255) / 256, 256, 0, stream>>>(P.sw2, Wb2, K2 * K2);
  cvt_bf16_kernel<<<(K2 * K2 + 255) / 256, 256, 0, stream>>>(P.sw3, Wb3, K2 * K2);
  copyq_kernel<<<(QSZ + 255) / 256, 256, 0, stream>>>(P.q, (float*)d_out);
  knn_kernel<<<dim3(M_ / 256, B_), 256, 0, stream>>>(P.p, P.q, idx);
  feat_kernel<<<NPTS / 8, 256, 0, stream>>>(P, idx, F, t1);
  gemm256_kernel<<<NPTS / 16, 128, 0, stream>>>(t1, Wb2, P.sb2, P.s2g, P.s2b,
                                                P.s2m, P.s2v, 1, t2);
  gemm256_kernel<<<NPTS / 16, 128, 0, stream>>>(t2, Wb3, P.sb3, P.s2g, P.s2b,
                                                P.s2m, P.s2v, 0, t3);
  xform_kernel<<<NPTS / 8, 256, 0, stream>>>(t3, F, Y);
  final_kernel<<<NPTS / 16, 128, 0, stream>>>(Y, Wbc, P, (float*)d_out);
}
